// NA_ALISTA_40553081209410
// MI455X (gfx1250) — matrix operand order repaired and checked by reading, compiled, not run on hardware
//
#include <hip/hip_runtime.h>
#include <hip/hip_bf16.h>
#include <math.h>

// ---------------------------------------------------------------------------
// Problem constants (from the reference)
// ---------------------------------------------------------------------------
#define BB   1024   // batch
#define MM   1024   // rows of A / cols of b
#define NN   4096   // cols of A / sparse code dim
#define KK   16     // iterations
#define DIM  128    // LSTM hidden
#define PP   100    // top-k

typedef __attribute__((ext_vector_type(16))) __bf16        v16bf;
typedef __attribute__((ext_vector_type(8)))  float         v8f;
typedef __attribute__((ext_vector_type(4)))  unsigned int  v4u;

union FragU { v4u u[2]; v16bf v; };

__device__ __forceinline__ unsigned short f2bf(float f) {
    unsigned int u = __float_as_uint(f);
    unsigned int r = u + 0x7fffu + ((u >> 16) & 1u);  // round-to-nearest-even
    return (unsigned short)(r >> 16);
}
__device__ __forceinline__ float sigmoidf(float x) { return 1.0f / (1.0f + __expf(-x)); }
__device__ __forceinline__ float softplusf(float x) {
    return (x > 20.0f) ? x : log1pf(expf(x));
}

// ---------------------------------------------------------------------------
// Elementwise f32 -> bf16 convert (used once for A)
// ---------------------------------------------------------------------------
__global__ void convert_bf16_kernel(const float* __restrict__ in,
                                    unsigned short* __restrict__ out, int n) {
    int stride = gridDim.x * blockDim.x;
    for (int i = blockIdx.x * blockDim.x + threadIdx.x; i < n; i += stride)
        out[i] = f2bf(in[i]);
}

// Wt[c][r] = bf16(W[r][c])   (W is rows x cols row-major)
__global__ void transpose_bf16_kernel(const float* __restrict__ W,
                                      unsigned short* __restrict__ Wt,
                                      int rows, int cols) {
    int stride = gridDim.x * blockDim.x;
    int total = rows * cols;
    for (int i = blockIdx.x * blockDim.x + threadIdx.x; i < total; i += stride) {
        int r = i / cols, c = i - r * cols;
        Wt[(size_t)c * rows + r] = f2bf(W[i]);
    }
}

// x = 0 (f32 and bf16 mirror), hidden[r][d] = h0[d], cell[r][d] = c0[d]
__global__ void init_state_kernel(float* __restrict__ x, unsigned short* __restrict__ xbf,
                                  float* __restrict__ hidden, float* __restrict__ cell,
                                  const float* __restrict__ h0, const float* __restrict__ c0) {
    int stride = gridDim.x * blockDim.x;
    int i0 = blockIdx.x * blockDim.x + threadIdx.x;
    for (int i = i0; i < BB * NN; i += stride) { x[i] = 0.0f; xbf[i] = 0; }
    for (int i = i0; i < BB * DIM; i += stride) {
        hidden[i] = h0[i & (DIM - 1)];
        cell[i]   = c0[i & (DIM - 1)];
    }
}

// ---------------------------------------------------------------------------
// bf16 WMMA GEMM:  Out[rows][cols] = X[rows][kd] * Wt[cols][kd]^T  (- Sub)
// Block = 256 threads = 8 wave32; block tile 256x128; per-wave tile 64x64
// (4x4 WMMA tiles -> 16 v_wmma per k-step vs 8 fragment ds_load_b128 pairs).
// ---------------------------------------------------------------------------
__global__ void __launch_bounds__(256)
gemm_bf16_wmma(const unsigned short* __restrict__ X,
               const unsigned short* __restrict__ Wt,
               float* __restrict__ Out,
               unsigned short* __restrict__ OutBf,   // optional bf16 mirror
               const float* __restrict__ Sub,        // optional elementwise subtract
               int rows, int cols, int kd) {
    __shared__ __attribute__((aligned(16))) unsigned short sX[256][40];
    __shared__ __attribute__((aligned(16))) unsigned short sW[128][40];

    const int tid  = threadIdx.x;
    const int lane = tid & 31;
    const int wv   = tid >> 5;
    const int wrow = (wv >> 1) * 64;      // 0,64,128,192
    const int wcol = (wv & 1) * 64;       // 0,64
    const int rowBase = blockIdx.y * 256;
    const int colBase = blockIdx.x * 128;
    const int h   = lane >> 4;            // half-wave
    const int l16 = lane & 15;

    v8f zero = {0.f, 0.f, 0.f, 0.f, 0.f, 0.f, 0.f, 0.f};
    v8f acc[4][4];
    for (int a = 0; a < 4; ++a)
        for (int b = 0; b < 4; ++b) acc[a][b] = zero;

    for (int k0 = 0; k0 < kd; k0 += 32) {
        // ---- stage 256x32 (X) and 128x32 (Wt) bf16 tiles into LDS ----
        #pragma unroll
        for (int i = 0; i < 4; ++i) {                  // 1024 chunks of 8 bf16
            int chunk = tid + 256 * i;
            int rr = chunk >> 2;
            int cc = (chunk & 3) * 8;
            *reinterpret_cast<v4u*>(&sX[rr][cc]) =
                *reinterpret_cast<const v4u*>(X + (size_t)(rowBase + rr) * kd + k0 + cc);
        }
        #pragma unroll
        for (int i = 0; i < 2; ++i) {                  // 512 chunks of 8 bf16
            int chunk = tid + 256 * i;
            int rr = chunk >> 2;
            int cc = (chunk & 3) * 8;
            *reinterpret_cast<v4u*>(&sW[rr][cc]) =
                *reinterpret_cast<const v4u*>(Wt + (size_t)(colBase + rr) * kd + k0 + cc);
        }
        if (k0 + 32 < kd) {   // global_prefetch_b8 of the next k-tile
            __builtin_prefetch(X  + (size_t)(rowBase + (tid >> 1)) * kd + k0 + 32, 0, 1);
            __builtin_prefetch(Wt + (size_t)(colBase + (tid >> 2)) * kd + k0 + 32, 0, 1);
        }
        __syncthreads();

        // ---- per-lane fragments (ISA 7.12.2 16-bit layouts), 16B ds loads ----
        FragU a[4], b[4];
        #pragma unroll
        for (int mi = 0; mi < 4; ++mi) {
            int m = wrow + mi * 16 + l16;
            a[mi].u[0] = *reinterpret_cast<const v4u*>(&sX[m][h * 8]);
            a[mi].u[1] = *reinterpret_cast<const v4u*>(&sX[m][16 + h * 8]);
        }
        #pragma unroll
        for (int ni = 0; ni < 4; ++ni) {
            int c = wcol + ni * 16 + l16;
            b[ni].u[0] = *reinterpret_cast<const v4u*>(&sW[c][h * 8]);
            b[ni].u[1] = *reinterpret_cast<const v4u*>(&sW[c][16 + h * 8]);
        }
        #pragma unroll
        for (int mi = 0; mi < 4; ++mi)
            #pragma unroll
            for (int ni = 0; ni < 4; ++ni)
                acc[mi][ni] = __builtin_amdgcn_wmma_f32_16x16x32_bf16(
                    false, a[mi].v, false, b[ni].v, (short)0, acc[mi][ni], false, false);
        __syncthreads();
    }

    // ---- epilogue: C/D layout -> row = base + r + 8*h, col = base + l16 ----
    #pragma unroll
    for (int mi = 0; mi < 4; ++mi)
        #pragma unroll
        for (int ni = 0; ni < 4; ++ni)
            #pragma unroll
            for (int rr = 0; rr < 8; ++rr) {
                int row = rowBase + wrow + mi * 16 + h * 8 + rr;
                int col = colBase + wcol + ni * 16 + l16;
                size_t idx = (size_t)row * cols + col;
                float v = acc[mi][ni][rr];
                if (Sub)   v -= Sub[idx];
                Out[idx] = v;
                if (OutBf) OutBf[idx] = f2bf(v);
            }
}

// ---------------------------------------------------------------------------
// Per-row L1 norm: out[r] = sum_j |in[r][j]|
// ---------------------------------------------------------------------------
__global__ void rowabs_kernel(const float* __restrict__ in, int len,
                              float* __restrict__ out) {
    __shared__ float red[256];
    int r = blockIdx.x;
    const float* row = in + (size_t)r * len;
    float s = 0.0f;
    for (int j = threadIdx.x; j < len; j += 256) s += fabsf(row[j]);
    red[threadIdx.x] = s;
    __syncthreads();
    for (int st = 128; st > 0; st >>= 1) {
        if (threadIdx.x < st) red[threadIdx.x] += red[threadIdx.x + st];
        __syncthreads();
    }
    if (threadIdx.x == 0) out[r] = red[0];
}

// mean / std(ddof=1) of bl1 and cl1 (single block, iter 0 only)
__global__ void stats_kernel(const float* __restrict__ bl1,
                             const float* __restrict__ cl1,
                             float* __restrict__ st) {
    __shared__ float s0[256], s1[256], s2[256], s3[256];
    float a = 0, b = 0, c = 0, d = 0;
    for (int i = threadIdx.x; i < BB; i += 256) {
        float u = bl1[i], v = cl1[i];
        a += u; b += u * u; c += v; d += v * v;
    }
    s0[threadIdx.x] = a; s1[threadIdx.x] = b; s2[threadIdx.x] = c; s3[threadIdx.x] = d;
    __syncthreads();
    for (int stp = 128; stp > 0; stp >>= 1) {
        if (threadIdx.x < stp) {
            s0[threadIdx.x] += s0[threadIdx.x + stp];
            s1[threadIdx.x] += s1[threadIdx.x + stp];
            s2[threadIdx.x] += s2[threadIdx.x + stp];
            s3[threadIdx.x] += s3[threadIdx.x + stp];
        }
        __syncthreads();
    }
    if (threadIdx.x == 0) {
        float n = (float)BB;
        float bm = s0[0] / n, cm = s2[0] / n;
        st[0] = bm;
        st[1] = sqrtf((s1[0] - n * bm * bm) / (n - 1.0f));
        st[2] = cm;
        st[3] = sqrtf((s3[0] - n * cm * cm) / (n - 1.0f));
    }
}

// ---------------------------------------------------------------------------
// LSTM cell + MLP head: one block (128 threads) per batch row
// ---------------------------------------------------------------------------
__global__ void __launch_bounds__(128)
lstm_kernel(const float* __restrict__ bl1, const float* __restrict__ cl1,
            const float* __restrict__ st,
            const float* __restrict__ Wih, const float* __restrict__ Whh,
            const float* __restrict__ bih, const float* __restrict__ bhh,
            const float* __restrict__ Wl,  const float* __restrict__ blv,
            const float* __restrict__ W2,  const float* __restrict__ b2,
            float* __restrict__ hidden, float* __restrict__ cell,
            float* __restrict__ gam, float* __restrict__ th,
            float* __restrict__ gam_out, float* __restrict__ th_out, int iter) {
    __shared__ float sH[DIM], sC[DIM], sR0[DIM], sR1[DIM];
    int r = blockIdx.x, j = threadIdx.x;
    sH[j] = hidden[(size_t)r * DIM + j];
    sC[j] = cell[(size_t)r * DIM + j];
    __syncthreads();

    float i0 = (bl1[r] - st[0]) / st[1];
    float i1 = (cl1[r] - st[2]) / st[3];

    float g[4];
    #pragma unroll
    for (int gg = 0; gg < 4; ++gg) {
        int idx = gg * DIM + j;
        float a = i0 * Wih[idx * 2] + i1 * Wih[idx * 2 + 1] + bih[idx] + bhh[idx];
        const float* wr = Whh + (size_t)idx * DIM;
        for (int d = 0; d < DIM; ++d) a += sH[d] * wr[d];
        g[gg] = a;
    }
    float cn = sigmoidf(g[1]) * sC[j] + sigmoidf(g[0]) * tanhf(g[2]);
    float hn = sigmoidf(g[3]) * tanhf(cn);
    __syncthreads();
    sC[j] = cn; sH[j] = hn;
    __syncthreads();
    cell[(size_t)r * DIM + j]   = cn;
    hidden[(size_t)r * DIM + j] = hn;

    float a = blv[j];
    const float* wr = Wl + (size_t)j * DIM;
    for (int d = 0; d < DIM; ++d) a += sC[d] * wr[d];
    float t = a > 0.0f ? a : 0.0f;
    sR0[j] = t * W2[j];
    sR1[j] = t * W2[DIM + j];
    __syncthreads();
    for (int stp = 64; stp > 0; stp >>= 1) {
        if (j < stp) { sR0[j] += sR0[j + stp]; sR1[j] += sR1[j + stp]; }
        __syncthreads();
    }
    if (j == 0) {
        float ga = softplusf(sR0[0] + b2[0]);
        float tt = softplusf(sR1[0] + b2[1]);
        gam[r] = ga; th[r] = tt;
        gam_out[(size_t)r * KK + iter] = ga;
        th_out[(size_t)r * KK + iter]  = tt;
    }
}

// ---------------------------------------------------------------------------
// d = x - 3*gamma*c ; exact P-th-largest |d| via binary search over IEEE bits;
// x = where(|d| > thresh, d, sign(d)*relu(|d| - theta/20));  also emits xbf.
// ---------------------------------------------------------------------------
__global__ void __launch_bounds__(256)
update_topk_kernel(float* __restrict__ x, unsigned short* __restrict__ xbf,
                   const float* __restrict__ c,
                   const float* __restrict__ gam, const float* __restrict__ th) {
    const int ELEMS = NN / 256;   // 16
    int r = blockIdx.x;
    const float g3  = 3.0f * gam[r];
    const float t20 = th[r] * (1.0f / 20.0f);
    size_t base = (size_t)r * NN;

    float        dv[ELEMS];
    unsigned int da[ELEMS];
    #pragma unroll
    for (int i = 0; i < ELEMS; ++i) {
        int j = i * 256 + threadIdx.x;
        float d = x[base + j] - g3 * c[base + j];
        dv[i] = d;
        da[i] = __float_as_uint(fabsf(d));
    }

    __shared__ int cnt;
    unsigned int lo = 0u, hi = 0x7f800000u;   // finite range
    while (lo < hi) {
        unsigned int mid = lo + ((hi - lo) >> 1);
        int local = 0;
        #pragma unroll
        for (int i = 0; i < ELEMS; ++i) local += (da[i] > mid) ? 1 : 0;
        if (threadIdx.x == 0) cnt = 0;
        __syncthreads();
        atomicAdd(&cnt, local);
        __syncthreads();
        int total = cnt;
        __syncthreads();
        if (total >= PP) lo = mid + 1; else hi = mid;
    }
    unsigned int thresh = lo;   // bits of P-th largest |d|

    #pragma unroll
    for (int i = 0; i < ELEMS; ++i) {
        int j = i * 256 + threadIdx.x;
        float out;
        if (da[i] > thresh) {
            out = dv[i];
        } else {
            float s = __uint_as_float(da[i]) - t20;
            s = s > 0.0f ? s : 0.0f;
            out = copysignf(s, dv[i]);
        }
        x[base + j]   = out;
        xbf[base + j] = f2bf(out);
    }
}

// ---------------------------------------------------------------------------
// Host orchestration
// ---------------------------------------------------------------------------
extern "C" void kernel_launch(void* const* d_in, const int* in_sizes, int n_in,
                              void* d_out, int out_size, void* d_ws, size_t ws_size,
                              hipStream_t stream) {
    const float* y   = (const float*)d_in[0];
    const float* A   = (const float*)d_in[1];
    const float* W   = (const float*)d_in[2];
    const float* Wih = (const float*)d_in[3];
    const float* Whh = (const float*)d_in[4];
    const float* bih = (const float*)d_in[5];
    const float* bhh = (const float*)d_in[6];
    const float* Wl  = (const float*)d_in[7];
    const float* blv = (const float*)d_in[8];
    const float* W2  = (const float*)d_in[9];
    const float* b2  = (const float*)d_in[10];
    const float* h0  = (const float*)d_in[11];
    const float* c0  = (const float*)d_in[12];

    float* x       = (float*)d_out;           // (B, N)
    float* gam_out = x + (size_t)BB * NN;     // (B, K)
    float* th_out  = gam_out + (size_t)BB * KK;

    char*  w   = (char*)d_ws;
    size_t off = 0;
    auto alloc = [&](size_t bytes) -> void* {
        void* p = w + off;
        off += (bytes + 255) & ~(size_t)255;
        return p;
    };
    unsigned short* Abf  = (unsigned short*)alloc((size_t)MM * NN * 2);
    unsigned short* Wtbf = (unsigned short*)alloc((size_t)NN * MM * 2);
    unsigned short* xbf  = (unsigned short*)alloc((size_t)BB * NN * 2);
    unsigned short* bbf  = (unsigned short*)alloc((size_t)BB * MM * 2);
    float* bf  = (float*)alloc((size_t)BB * MM * 4);
    float* cf  = (float*)alloc((size_t)BB * NN * 4);
    float* bl1 = (float*)alloc(BB * 4);
    float* cl1 = (float*)alloc(BB * 4);
    float* st  = (float*)alloc(64);
    float* hid = (float*)alloc((size_t)BB * DIM * 4);
    float* cel = (float*)alloc((size_t)BB * DIM * 4);
    float* gam = (float*)alloc(BB * 4);
    float* th  = (float*)alloc(BB * 4);

    // one-time prep
    convert_bf16_kernel<<<4096, 256, 0, stream>>>(A, Abf, MM * NN);
    transpose_bf16_kernel<<<4096, 256, 0, stream>>>(W, Wtbf, MM, NN);
    init_state_kernel<<<2048, 256, 0, stream>>>(x, xbf, hid, cel, h0, c0);

    dim3 g1(MM / 128, BB / 256);   // b = x A^T - y
    dim3 g2(NN / 128, BB / 256);   // c = b W

    for (int it = 0; it < KK; ++it) {
        gemm_bf16_wmma<<<g1, 256, 0, stream>>>(xbf, Abf, bf, bbf, y, BB, MM, NN);
        rowabs_kernel<<<BB, 256, 0, stream>>>(bf, MM, bl1);
        gemm_bf16_wmma<<<g2, 256, 0, stream>>>(bbf, Wtbf, cf, nullptr, nullptr, BB, NN, MM);
        rowabs_kernel<<<BB, 256, 0, stream>>>(cf, NN, cl1);
        if (it == 0) stats_kernel<<<1, 256, 0, stream>>>(bl1, cl1, st);
        lstm_kernel<<<BB, 128, 0, stream>>>(bl1, cl1, st, Wih, Whh, bih, bhh,
                                            Wl, blv, W2, b2, hid, cel,
                                            gam, th, gam_out, th_out, it);
        update_topk_kernel<<<BB, 256, 0, stream>>>(x, xbf, cf, gam, th);
    }
}
